// GAT_53111565582838
// MI455X (gfx1250) — matrix operand order repaired and checked by reading, not compiled, not run on hardware
//
#include <hip/hip_runtime.h>
#include <hip/hip_bf16.h>

typedef __attribute__((ext_vector_type(16))) __bf16 v16bf;
typedef __attribute__((ext_vector_type(8)))  __bf16 v8bf;
typedef __attribute__((ext_vector_type(8)))  float  v8f;

#define N_NODES 2048
#define N_EDGES 4096
#define F_INN   256
#define NHEAD   8
#define DHID    64
#define HDIM    512
#define SLOPE   0.2f
#define NEGV    (-9e16f)

#if __has_builtin(__builtin_amdgcn_global_load_async_to_lds_b32) && \
    __has_builtin(__builtin_amdgcn_s_wait_asynccnt)
#define HAVE_ASYNC_LDS 1
typedef __attribute__((address_space(1))) int as1_int;
typedef __attribute__((address_space(3))) int as3_int;
#else
#define HAVE_ASYNC_LDS 0
#endif

// ---------------------------------------------------------------------------
// f32 -> bf16 elementwise convert
// ---------------------------------------------------------------------------
__global__ void gat_cvt_bf16(const float* __restrict__ x, __bf16* __restrict__ y, int n) {
    int i = blockIdx.x * 256 + threadIdx.x;
    if (i < n) y[i] = (__bf16)x[i];
}

// W [256,512] f32 -> Wt [512,256] bf16 (transpose so B fragments load contiguously)
__global__ void gat_transpose_w(const float* __restrict__ w, __bf16* __restrict__ wt) {
    int j = blockIdx.x * 256 + threadIdx.x;   // j < 512*256
    int k = j & 255, c = j >> 8;
    wt[c * F_INN + k] = (__bf16)w[k * HDIM + c];
}

// ---------------------------------------------------------------------------
// GEMM: htT[f][m] = sum_k X[m][k] * W[k][f]   (bf16 WMMA, f32 accum)
// X row-major bf16 [M,256]; Wt transposed bf16 [512,256]; out htT bf16 [512,M]
// block = 256 thr = 8 waves, wave w -> feature cols [64w,64w+64), grid = M/16
// ---------------------------------------------------------------------------
__global__ __launch_bounds__(256) void gat_gemm_wmma(
    const __bf16* __restrict__ xbf, const __bf16* __restrict__ wt,
    __bf16* __restrict__ htT, int M) {
    const int lane = threadIdx.x & 31;
    const int wave = threadIdx.x >> 5;
    const int n    = lane & 15;
    const int hs   = lane >> 4;
    const int k0   = 8 * hs;     // A-fragment K base per lane half
    const int kb   = 16 * hs;    // unused: B fragment also uses k0
    const int row16 = blockIdx.x * 16;
    const int grow  = row16 + n;

    v8f acc[4] = {v8f{}, v8f{}, v8f{}, v8f{}};

    for (int kk = 0; kk < F_INN; kk += 32) {
        // A fragment: 16x32 bf16, lane row = lane%16, two contiguous 16B loads
        const __bf16* ap = xbf + (size_t)grow * F_INN + kk + k0;
        v8bf a0 = *(const v8bf*)ap;
        v8bf a1 = *(const v8bf*)(ap + 16);
        v16bf af;
#pragma unroll
        for (int e = 0; e < 8; ++e) { af[e] = a0[e]; af[e + 8] = a1[e]; }

#pragma unroll
        for (int t = 0; t < 4; ++t) {
            const int f = wave * 64 + t * 16 + n;
            const __bf16* bp = wt + (size_t)f * F_INN + kk + k0;
            v8bf b0 = *(const v8bf*)bp;
            v8bf b1 = *(const v8bf*)(bp + 16);
            v16bf bf;
#pragma unroll
            for (int e = 0; e < 8; ++e) { bf[e] = b0[e]; bf[e + 8] = b1[e]; }
            acc[t] = __builtin_amdgcn_wmma_f32_16x16x32_bf16(
                false, af, false, bf, (short)0, acc[t], false, false);
        }
    }
    // epilogue: write transposed bf16 (rows contiguous -> one 16B store per tile)
#pragma unroll
    for (int t = 0; t < 4; ++t) {
        const int f = wave * 64 + t * 16 + n;
        v8bf o;
#pragma unroll
        for (int e = 0; e < 8; ++e) o[e] = (__bf16)acc[t][e];
        *(v8bf*)(htT + (size_t)f * M + row16 + 8 * hs) = o;
    }
}

// ---------------------------------------------------------------------------
// src/tgt attention scalars: out[h,i] = sum_d htT[h*64+d][i] * a[h*128+half+d]
// ---------------------------------------------------------------------------
__global__ void gat_dots(const __bf16* __restrict__ htT, const float* __restrict__ a,
                         int M, int halfofs, float* __restrict__ out) {
    int id = blockIdx.x * 256 + threadIdx.x;
    if (id >= NHEAD * M) return;
    int h = id / M, i = id - h * M;
    float s = 0.f;
#pragma unroll 8
    for (int d = 0; d < DHID; ++d)
        s += (float)htT[(size_t)(h * DHID + d) * M + i] * a[h * 2 * DHID + halfofs + d];
    out[h * M + i] = s;
}

// ---------------------------------------------------------------------------
// Tile staging: 16x32 mask (int) + 8x32 tgt (float) per column tile,
// double-buffered; async global->LDS on gfx1250 when available.
// ---------------------------------------------------------------------------
__device__ __forceinline__ void tile_issue(int* mdst, float* tdst,
                                           const int* __restrict__ mat,
                                           const float* __restrict__ tgt,
                                           int row16, int Mcols, int cb) {
    // 512 mask ints: 2 per thread
    int idx = threadIdx.x;
#pragma unroll
    for (int rep = 0; rep < 2; ++rep, idx += 256) {
        const int r = idx >> 5, c = idx & 31;
        const int* g = &mat[(size_t)(row16 + r) * Mcols + cb + c];
#if HAVE_ASYNC_LDS
        __builtin_amdgcn_global_load_async_to_lds_b32(
            (as1_int*)g, (as3_int*)&mdst[idx], 0, 0);
#else
        mdst[idx] = __builtin_nontemporal_load(g);
#endif
    }
    // 256 tgt floats (all 8 heads x 32 cols): 1 per thread
    {
        const int h2 = threadIdx.x >> 5, c = threadIdx.x & 31;
        const float* g = &tgt[(size_t)h2 * Mcols + cb + c];
#if HAVE_ASYNC_LDS
        __builtin_amdgcn_global_load_async_to_lds_b32(
            (as1_int*)g, (as3_int*)&tdst[threadIdx.x], 0, 0);
#else
        tdst[threadIdx.x] = *g;
#endif
    }
}

__device__ __forceinline__ void tile_wait() {
#if HAVE_ASYNC_LDS
    __builtin_amdgcn_s_wait_asynccnt(0);
#endif
    __syncthreads();
}

// ---------------------------------------------------------------------------
// Fused flash-style GAT aggregation.
// grid = N/16 row tiles; block = 256 thr = 8 waves; wave h handles head h.
// Mask+tgt tiles shared in LDS across all 8 heads (read once, not 8x),
// double-buffered with async global->LDS copies overlapping WMMA compute.
// Online softmax in f32; P tile lands directly in bf16 A-fragment layout;
// V fragments are contiguous 16B loads from transposed bf16 ht (L2 resident).
// ---------------------------------------------------------------------------
__global__ __launch_bounds__(256) void gat_flash(
    const __bf16* __restrict__ htTn, const __bf16* __restrict__ htTe,
    const float* __restrict__ srcN, const float* __restrict__ tgtN,
    const float* __restrict__ srcE, const float* __restrict__ tgtE,
    const int* __restrict__ matN, const int* __restrict__ matE,
    float* __restrict__ hpart) {
    __shared__ int   maskTile[2][16 * 32];
    __shared__ float tgtTile[2][8 * 32];

    const int lane  = threadIdx.x & 31;
    const int h     = threadIdx.x >> 5;   // wave id == head id
    const int n     = lane & 15;
    const int hs    = lane >> 4;
    const int k0    = 8 * hs;
    const int kb    = 16 * hs;
    const int row16 = blockIdx.x * 16;
    const int grow  = row16 + n;

    v8f outacc[4] = {v8f{}, v8f{}, v8f{}, v8f{}};

    for (int phase = 0; phase < 2; ++phase) {
        const int M           = phase ? N_EDGES : N_NODES;
        const __bf16* V       = phase ? htTe : htTn;
        const float* src      = phase ? srcE : srcN;
        const float* tgt      = phase ? tgtE : tgtN;
        const int* __restrict__ mat = phase ? matE : matN;

        const float srow = src[h * N_NODES + grow];
        float mrun = -3.0e38f, lrun = 0.f;
        v8f acc[4] = {v8f{}, v8f{}, v8f{}, v8f{}};

        int p = 0;
        tile_issue(maskTile[0], tgtTile[0], mat, tgt, row16, M, 0);
        tile_wait();

        for (int cb = 0; cb < M; cb += 32) {
            if (cb + 32 < M)   // async-prefetch next tile into other buffer
                tile_issue(maskTile[p ^ 1], tgtTile[p ^ 1], mat, tgt, row16, M, cb + 32);
            const int*   mk = maskTile[p];
            const float* tk = tgtTile[p];

            // --- scores for my row, my 16 K columns (A-fragment ordering) ---
            float p16[16];
            float lmax = -3.0e38f;
#pragma unroll
            for (int e = 0; e < 16; ++e) {
                const int K = (e < 8) ? (k0 + e) : (k0 + e + 8);
                float ev = srow + tk[h * 32 + K];
                ev = (ev >= 0.f) ? ev : SLOPE * ev;
                if (mk[n * 32 + K] == 0) ev = NEGV;
                p16[e] = ev;
                lmax = fmaxf(lmax, ev);
            }
            const float rmax = fmaxf(lmax, __shfl_xor(lmax, 16, 32));
            const float mnew = fmaxf(mrun, rmax);
            const float corr = __expf(mrun - mnew);
            float lsum = 0.f;
            v16bf af;
#pragma unroll
            for (int e = 0; e < 16; ++e) {
                const float pe = __expf(p16[e] - mnew);
                lsum += pe;
                af[e] = (__bf16)pe;
            }
            lsum += __shfl_xor(lsum, 16, 32);
            lrun = lrun * corr + lsum;
            mrun = mnew;

            // rescale accumulators: row of acc VGPR vg is (vg + 8*hs)
#pragma unroll
            for (int vg = 0; vg < 8; ++vg) {
                const float c = __shfl(corr, vg + 8 * hs, 32);
                acc[0][vg] *= c; acc[1][vg] *= c; acc[2][vg] *= c; acc[3][vg] *= c;
            }

            // --- P(16x32) @ V(32x64): 4 WMMAs, V via contiguous 16B loads ---
#pragma unroll
            for (int t = 0; t < 4; ++t) {
                const __bf16* bp = V + (size_t)(h * DHID + t * 16 + n) * M + cb + k0;
                v8bf b0 = *(const v8bf*)bp;
                v8bf b1 = *(const v8bf*)(bp + 16);
                v16bf bf;
#pragma unroll
                for (int e = 0; e < 8; ++e) { bf[e] = b0[e]; bf[e + 8] = b1[e]; }
                acc[t] = __builtin_amdgcn_wmma_f32_16x16x32_bf16(
                    false, af, false, bf, (short)0, acc[t], false, false);
            }

            tile_wait();   // next tile resident + everyone done with this one
            p ^= 1;
        }

        // finalize this softmax: out += acc / l
        const float linv = 1.0f / lrun;
#pragma unroll
        for (int vg = 0; vg < 8; ++vg) {
            const float s = __shfl(linv, vg + 8 * hs, 32);
            outacc[0][vg] += acc[0][vg] * s;
            outacc[1][vg] += acc[1][vg] * s;
            outacc[2][vg] += acc[2][vg] * s;
            outacc[3][vg] += acc[3][vg] * s;
        }
    }

    // store per-head partial h' (deterministic; reduced later)
#pragma unroll
    for (int t = 0; t < 4; ++t)
#pragma unroll
        for (int vg = 0; vg < 8; ++vg)
            hpart[((size_t)h * N_NODES + row16 + vg + 8 * hs) * DHID +
                  t * 16 + n] = outacc[t][vg];
}

// mean over heads
__global__ void gat_reduce_mean(const float* __restrict__ hpart, float* __restrict__ out) {
    int i = blockIdx.x * 256 + threadIdx.x;   // < 2048*64
    float s = 0.f;
#pragma unroll
    for (int h = 0; h < NHEAD; ++h) s += hpart[(size_t)h * N_NODES * DHID + i];
    out[i] = s * (1.0f / NHEAD);
}

// ---------------------------------------------------------------------------
extern "C" void kernel_launch(void* const* d_in, const int* in_sizes, int n_in,
                              void* d_out, int out_size, void* d_ws, size_t ws_size,
                              hipStream_t stream) {
    const float* nodes = (const float*)d_in[0];   // [2048,256]
    const float* edges = (const float*)d_in[1];   // [4096,256]
    const float* WN    = (const float*)d_in[2];   // [256,512]
    const float* WE    = (const float*)d_in[3];   // [256,512]
    const float* aN    = (const float*)d_in[4];   // [8,128,1]
    const float* aE    = (const float*)d_in[5];   // [8,128,1]
    const int*   matN  = (const int*)d_in[6];     // [2048,2048]
    const int*   matE  = (const int*)d_in[7];     // [2048,4096]
    float* out = (float*)d_out;                   // [2048,64]

    char* ws = (char*)d_ws;
    size_t off = 0;
    auto alloc = [&](size_t bytes) { void* p = ws + off; off += (bytes + 255) & ~(size_t)255; return p; };
    __bf16* xn_bf = (__bf16*)alloc((size_t)N_NODES * F_INN * 2);
    __bf16* xe_bf = (__bf16*)alloc((size_t)N_EDGES * F_INN * 2);
    __bf16* wtN   = (__bf16*)alloc((size_t)HDIM * F_INN * 2);
    __bf16* wtE   = (__bf16*)alloc((size_t)HDIM * F_INN * 2);
    __bf16* htTn  = (__bf16*)alloc((size_t)HDIM * N_NODES * 2);
    __bf16* htTe  = (__bf16*)alloc((size_t)HDIM * N_EDGES * 2);
    float*  srcN  = (float*)alloc((size_t)NHEAD * N_NODES * 4);
    float*  tgtN  = (float*)alloc((size_t)NHEAD * N_NODES * 4);
    float*  srcE  = (float*)alloc((size_t)NHEAD * N_NODES * 4);
    float*  tgtE  = (float*)alloc((size_t)NHEAD * N_EDGES * 4);
    float*  hpart = (float*)alloc((size_t)NHEAD * N_NODES * DHID * 4);

    // 1) precision convert + W transpose
    gat_cvt_bf16<<<(N_NODES * F_INN) / 256, 256, 0, stream>>>(nodes, xn_bf, N_NODES * F_INN);
    gat_cvt_bf16<<<(N_EDGES * F_INN) / 256, 256, 0, stream>>>(edges, xe_bf, N_EDGES * F_INN);
    gat_transpose_w<<<(HDIM * F_INN) / 256, 256, 0, stream>>>(WN, wtN);
    gat_transpose_w<<<(HDIM * F_INN) / 256, 256, 0, stream>>>(WE, wtE);

    // 2) projection GEMMs -> transposed bf16 ht
    gat_gemm_wmma<<<N_NODES / 16, 256, 0, stream>>>(xn_bf, wtN, htTn, N_NODES);
    gat_gemm_wmma<<<N_EDGES / 16, 256, 0, stream>>>(xe_bf, wtE, htTe, N_EDGES);

    // 3) attention score scalars
    gat_dots<<<(NHEAD * N_NODES) / 256, 256, 0, stream>>>(htTn, aN, N_NODES, 0,    srcN);
    gat_dots<<<(NHEAD * N_NODES) / 256, 256, 0, stream>>>(htTn, aN, N_NODES, DHID, tgtN);
    gat_dots<<<(NHEAD * N_NODES) / 256, 256, 0, stream>>>(htTn, aE, N_NODES, 0,    srcE);
    gat_dots<<<(NHEAD * N_EDGES) / 256, 256, 0, stream>>>(htTe, aE, N_EDGES, DHID, tgtE);

    // 4) fused masked-softmax attention + PV aggregation (flash style)
    gat_flash<<<N_NODES / 16, 256, 0, stream>>>(htTn, htTe, srcN, tgtN, srcE, tgtE,
                                                matN, matE, hpart);

    // 5) mean over heads
    gat_reduce_mean<<<(N_NODES * DHID) / 256, 256, 0, stream>>>(hpart, out);
}
